// DynamicsNet_489626271748
// MI455X (gfx1250) — matrix operand order repaired and checked by reading, not compiled, not run on hardware
//
#include <hip/hip_runtime.h>
#include <hip/hip_bf16.h>

typedef __attribute__((ext_vector_type(16))) _Float16 v16h;
typedef __attribute__((ext_vector_type(8)))  _Float16 v8h;
typedef __attribute__((ext_vector_type(8)))  float    v8f;

#define N_NODES 96
#define D_MODEL 128
#define BN      (N_NODES * D_MODEL)   // 12288 elems per sample

// ---- f16 weight workspace layout (in halves) ----
#define WS_A16_OFF   0
#define WS_PW16_OFF  (96 * 96)                    // 9216
#define WS_CW16_OFF  (WS_PW16_OFF + 128 * 256)    // 41984
#define WS_TOTAL_H   (WS_CW16_OFF + 8 * 2 * 128 * 128)  // 304128 halves

// ---- LDS layout (bytes) ----
// region0: union of xin16 [96][264] f16 (50688 B) and y32 [96][136] f32 (52224 B)
#define SM_REGION0   0
#define SM_X32       52224                   // x32 [96][132] f32 : 50688 B
#define SM_XT16      (SM_X32 + 50688)        // xT16 [128][104] f16 : 26624 B
#define SM_MSG16     (SM_XT16 + 26624)       // msg16 [96][136] f16 : 26112 B
#define SM_MISC      (SM_MSG16 + 26112)      // misc floats
#define SMEM_BYTES   (SM_MISC + 3616)

#define XIN_RS 264
#define Y32_RS 136
#define X32_RS 132
#define XT_RS  104
#define MSG_RS 136

// A operand tile (16x32 f16) from [M][K] row-major storage.
// ISA 16-bit A layout: lanes 0-15 -> K {0..7,16..23}, lanes 16-31 -> K {8..15,24..31}.
__device__ __forceinline__ v16h load_a_tile(const _Float16* base, int rs,
                                            int m0, int k0, int lane) {
  int m  = m0 + (lane & 15);
  int kb = k0 + ((lane & 16) >> 1);   // +8 for upper half-wave
  const _Float16* p = base + m * rs + kb;
  v8h lo = *(const v8h*)(p);          // K = kb .. kb+7
  v8h hi = *(const v8h*)(p + 16);     // K = kb+16 .. kb+23
  v16h r;
#pragma unroll
  for (int i = 0; i < 8; ++i) { r[i] = lo[i]; r[i + 8] = hi[i]; }
  return r;
}

// B operand tile (32x16 f16) from [N][K] storage (K contiguous per N).
// ISA B layout: lanes 0-15 hold K {0..7,16..23}, lanes 16-31 hold K {8..15,24..31}.
__device__ __forceinline__ v16h load_b_tile(const _Float16* base, int rs,
                                            int n0, int k0, int lane) {
  int n  = n0 + (lane & 15);
  int kb = k0 + ((lane & 16) >> 1);   // +8 for upper half-wave
  const _Float16* p = base + n * rs + kb;
  v8h lo = *(const v8h*)(p);
  v8h hi = *(const v8h*)(p + 16);
  v16h r;
#pragma unroll
  for (int i = 0; i < 8; ++i) { r[i] = lo[i]; r[i + 8] = hi[i]; }
  return r;
}

// M=96 x N=16 (this wave's column tile) x K=32*KTILES matmul into 6 f32 accumulators.
template <int KTILES>
__device__ __forceinline__ void mm_tiles(const _Float16* A, int ars,
                                         const _Float16* Bm, int brs,
                                         int wv, int lane, v8f (&acc)[6]) {
  v8f z = {};
#pragma unroll
  for (int mt = 0; mt < 6; ++mt) acc[mt] = z;
  for (int kt = 0; kt < KTILES; ++kt) {
    v16h bt = load_b_tile(Bm, brs, wv * 16, kt * 32, lane);
#pragma unroll
    for (int mt = 0; mt < 6; ++mt) {
      v16h at = load_a_tile(A, ars, mt * 16, kt * 32, lane);
      acc[mt] = __builtin_amdgcn_wmma_f32_16x16x32_f16(
          false, at, false, bt, (short)0, acc[mt], false, false);
    }
  }
}

__device__ __forceinline__ float mishf(float x) {
  float sp = (x > 20.0f) ? x : log1pf(__expf(x));
  return x * tanhf(sp);
}

// ---------- prep: convert weights to f16 into workspace ----------
__global__ __launch_bounds__(256) void dynnet_prep_f16(
    const float* __restrict__ A_norm, const float* __restrict__ proj_w,
    const float* __restrict__ conv_w, _Float16* __restrict__ ws) {
  int i = blockIdx.x * 256 + threadIdx.x;
  if (i < WS_PW16_OFF) {
    ws[i] = (_Float16)A_norm[i];                    // [96][96], symmetric
  } else if (i < WS_CW16_OFF) {
    ws[i] = (_Float16)proj_w[i - WS_PW16_OFF];      // [128][256], K-contig
  } else if (i < WS_TOTAL_H) {
    ws[i] = (_Float16)conv_w[i - WS_CW16_OFF];      // [8][2][128][128], K-contig
  }
}

// ---------- main: one workgroup (8 wave32) per batch sample ----------
__global__ __launch_bounds__(256) void dynnet_main(
    const float* __restrict__ h, const int* __restrict__ a,
    const int* __restrict__ piece_id,
    const float* __restrict__ piece_emb, const float* __restrict__ pos_emb,
    const float* __restrict__ proj_b, const float* __restrict__ conv_b,
    const float* __restrict__ ln_g, const float* __restrict__ ln_b,
    const float* __restrict__ r1_w, const float* __restrict__ r1_b,
    const float* __restrict__ rn_g, const float* __restrict__ rn_b,
    const float* __restrict__ r2_w, const float* __restrict__ r2_b,
    const _Float16* __restrict__ ws16, float* __restrict__ out, int B) {
  extern __shared__ __align__(16) char smem_raw[];
  _Float16* xin16 = (_Float16*)(smem_raw + SM_REGION0);  // [96][264]
  float*    y32   = (float*)   (smem_raw + SM_REGION0);  // [96][136] (aliases xin16)
  float*    x32   = (float*)   (smem_raw + SM_X32);      // [96][132]
  _Float16* xT16  = (_Float16*)(smem_raw + SM_XT16);     // [128][104]
  _Float16* msg16 = (_Float16*)(smem_raw + SM_MSG16);    // [96][136]
  float* a_emb = (float*)(smem_raw + SM_MISC);           // 128
  float* hmean = a_emb + 128;                            // 128
  float* rbuf  = hmean + 128;                            // 64
  float* ract  = rbuf + 64;                              // 64
  float* redA  = ract + 64;                              // 256
  float* redB  = redA + 256;                             // 256
  float* stats = redB + 256;                             // 4

  const _Float16* A16  = ws16 + WS_A16_OFF;   // [96][96]
  const _Float16* PW16 = ws16 + WS_PW16_OFF;  // [128][256]
  const _Float16* CW16 = ws16 + WS_CW16_OFF;  // [16][128][128]

  const int b = blockIdx.x;
  const int t = threadIdx.x;
  const int wv = t >> 5, lane = t & 31;
  const float* hb = h + (size_t)b * BN;

  // ---- phase 0: action embedding + h transpose into xin16 [n][c] ----
  if (t < 128) {
    int pidb = piece_id[b];
    int node = (int)(((unsigned)a[b]) % 96u);
    float ae = piece_emb[pidb * 128 + t] + pos_emb[node * 128 + t];
    a_emb[t] = ae;
    _Float16 ah = (_Float16)ae;
    for (int n = 0; n < 96; ++n) xin16[n * XIN_RS + 128 + t] = ah;
  }
  for (int idx = t; idx < BN; idx += 256) {        // h[b] is [c][n]; coalesced read
    int c = idx / 96;
    int n = idx - c * 96;
    xin16[n * XIN_RS + c] = (_Float16)hb[idx];
  }
  __syncthreads();

  // ---- reward head (tiny) ----
  if (t < 128) {
    float s = 0.f;
    for (int n = 0; n < 96; ++n) s += (float)xin16[n * XIN_RS + t];
    hmean[t] = s * (1.0f / 96.0f);
  }
  __syncthreads();
  if (t < 64) {
    const float* wr = r1_w + t * 256;
    float s = r1_b[t];
    for (int c = 0; c < 128; ++c) s += hmean[c] * wr[c];
    for (int c = 0; c < 128; ++c) s += a_emb[c] * wr[128 + c];
    rbuf[t] = s;
  }
  __syncthreads();
  if (t == 0) {
    float m = 0.f;
    for (int k = 0; k < 64; ++k) m += rbuf[k];
    m *= (1.0f / 64.0f);
    float v = 0.f;
    for (int k = 0; k < 64; ++k) { float d = rbuf[k] - m; v += d * d; }
    v *= (1.0f / 64.0f);
    stats[0] = m; stats[1] = rsqrtf(v + 1e-5f);
  }
  __syncthreads();
  if (t < 64) {
    float xv = (rbuf[t] - stats[0]) * stats[1] * rn_g[t] + rn_b[t];
    ract[t] = mishf(xv);
  }
  __syncthreads();
  {
    size_t loff = (size_t)B * BN + (size_t)b * 401;
    for (int o = t; o < 401; o += 256) {
      const float* wr = r2_w + o * 64;
      float s = r2_b[o];
      for (int k = 0; k < 64; ++k) s += ract[k] * wr[k];
      out[loff + o] = s;
    }
  }

  // ---- proj: x[n][d] = xin[96x256] @ proj_w^T, +bias ----
  {
    v8f acc[6];
    mm_tiles<8>(xin16, XIN_RS, PW16, 256, wv, lane, acc);
    int e = wv * 16 + (lane & 15);
    float pb = proj_b[e];
#pragma unroll
    for (int mt = 0; mt < 6; ++mt) {
      int nb = mt * 16 + ((lane & 16) >> 1);
#pragma unroll
      for (int v = 0; v < 8; ++v) {
        float val = acc[mt][v] + pb;
        x32[(nb + v) * X32_RS + e] = val;
        xT16[e * XT_RS + nb + v] = (_Float16)val;
      }
    }
  }
  __syncthreads();

  // ---- 8 residual blocks, 2 gconvs each ----
  for (int blk = 0; blk < 8; ++blk) {
    for (int g = 0; g < 2; ++g) {
      // msg[i][d] = A_norm @ cur  (A operand: A16 rows; B operand: xT16 [d][j])
      {
        v8f acc[6];
        mm_tiles<3>(A16, 96, xT16, XT_RS, wv, lane, acc);
        int d = wv * 16 + (lane & 15);
#pragma unroll
        for (int mt = 0; mt < 6; ++mt) {
          int ib = mt * 16 + ((lane & 16) >> 1);
#pragma unroll
          for (int v = 0; v < 8; ++v)
            msg16[(ib + v) * MSG_RS + d] = (_Float16)acc[mt][v];
        }
      }
      __syncthreads();
      // y[n][e] = msg @ w^T  (A operand: msg16 rows; B operand: conv_w rows)
      {
        const _Float16* W = CW16 + (size_t)(blk * 2 + g) * 16384;
        v8f acc[6];
        mm_tiles<4>(msg16, MSG_RS, W, 128, wv, lane, acc);
        int e = wv * 16 + (lane & 15);
#pragma unroll
        for (int mt = 0; mt < 6; ++mt) {
          int nb = mt * 16 + ((lane & 16) >> 1);
#pragma unroll
          for (int v = 0; v < 8; ++v)
            y32[(nb + v) * Y32_RS + e] = acc[mt][v];
        }
      }
      __syncthreads();
      // bias + layernorm (+mish on g==0, +residual on g==1); refresh xT16 staging
      {
        const float* cb = conv_b + (blk * 2 + g) * 128;
        const float* lg = ln_g + (blk * 2 + g) * 128;
        const float* lb = ln_b + (blk * 2 + g) * 128;
        if (t < 96) {
          float* row = y32 + t * Y32_RS;
          float m = 0.f;
          for (int e = 0; e < 128; ++e) m += row[e] + cb[e];
          m *= (1.0f / 128.0f);
          float var = 0.f;
          for (int e = 0; e < 128; ++e) {
            float u = row[e] + cb[e] - m;
            var += u * u;
          }
          float inv = rsqrtf(var * (1.0f / 128.0f) + 1e-5f);
          if (g == 0) {
            for (int e = 0; e < 128; ++e) {
              float u = (row[e] + cb[e] - m) * inv * lg[e] + lb[e];
              xT16[e * XT_RS + t] = (_Float16)mishf(u);
            }
          } else {
            for (int e = 0; e < 128; ++e) {
              float u = (row[e] + cb[e] - m) * inv * lg[e] + lb[e];
              float nx = x32[t * X32_RS + e] + u;
              x32[t * X32_RS + e] = nx;
              xT16[e * XT_RS + t] = (_Float16)nx;
            }
          }
        }
      }
      __syncthreads();
    }
  }

  // ---- per-sample min-max rescale, write h_scaled [d][n] ----
  float mn = 3.4e38f, mx = -3.4e38f;
  for (int idx = t; idx < BN; idx += 256) {
    int n = idx >> 7, d = idx & 127;
    float v = x32[n * X32_RS + d];
    mn = fminf(mn, v);
    mx = fmaxf(mx, v);
  }
  redA[t] = mn; redB[t] = mx;
  __syncthreads();
  for (int s = 128; s > 0; s >>= 1) {
    if (t < s) {
      redA[t] = fminf(redA[t], redA[t + s]);
      redB[t] = fmaxf(redB[t], redB[t + s]);
    }
    __syncthreads();
  }
  float gmn = redA[0];
  float scale = redB[0] - gmn;
  if (scale < 1e-5f) scale += 1e-5f;
  float inv = 1.0f / scale;
  float* ob = out + (size_t)b * BN;
  for (int idx = t; idx < BN; idx += 256) {     // idx = d*96+n, coalesced store
    int d = idx / 96;
    int n = idx - d * 96;
    ob[idx] = (x32[n * X32_RS + d] - gmn) * inv;
  }
}

extern "C" void kernel_launch(void* const* d_in, const int* in_sizes, int n_in,
                              void* d_out, int out_size, void* d_ws, size_t ws_size,
                              hipStream_t stream) {
  const float* h         = (const float*)d_in[0];
  const int*   a         = (const int*)  d_in[1];
  const int*   piece_id  = (const int*)  d_in[2];
  const float* A_norm    = (const float*)d_in[3];
  const float* piece_emb = (const float*)d_in[4];
  const float* pos_emb   = (const float*)d_in[5];
  const float* proj_w    = (const float*)d_in[6];
  const float* proj_b    = (const float*)d_in[7];
  const float* conv_w    = (const float*)d_in[8];
  const float* conv_b    = (const float*)d_in[9];
  const float* ln_g      = (const float*)d_in[10];
  const float* ln_b      = (const float*)d_in[11];
  const float* r1_w      = (const float*)d_in[12];
  const float* r1_b      = (const float*)d_in[13];
  const float* rn_g      = (const float*)d_in[14];
  const float* rn_b      = (const float*)d_in[15];
  const float* r2_w      = (const float*)d_in[16];
  const float* r2_b      = (const float*)d_in[17];
  float* out = (float*)d_out;
  _Float16* ws16 = (_Float16*)d_ws;
  int B = in_sizes[1];  // 2048

  (void)hipFuncSetAttribute((const void*)dynnet_main,
                            hipFuncAttributeMaxDynamicSharedMemorySize, SMEM_BYTES);

  dynnet_prep_f16<<<(WS_TOTAL_H + 255) / 256, 256, 0, stream>>>(
      A_norm, proj_w, conv_w, ws16);
  dynnet_main<<<B, 256, SMEM_BYTES, stream>>>(
      h, a, piece_id, piece_emb, pos_emb, proj_b, conv_b, ln_g, ln_b,
      r1_w, r1_b, rn_g, rn_b, r2_w, r2_b, ws16, out, B);
}
